// edge_layer_87832081203484
// MI455X (gfx1250) — matrix operand order repaired and checked by reading, not compiled, not run on hardware
//
#include <hip/hip_runtime.h>

typedef __attribute__((ext_vector_type(16))) _Float16 v16h;
typedef __attribute__((ext_vector_type(8)))  _Float16 v8h;
typedef __attribute__((ext_vector_type(8)))  float    v8f;
typedef unsigned int u32x4 __attribute__((ext_vector_type(4)));
typedef int          i32x8 __attribute__((ext_vector_type(8)));
typedef int          i32x4 __attribute__((ext_vector_type(4)));

#define N_TOK 9216      // 96*96 tokens
#define EMB   128
#define NT    576       // 16x16 tiles per dim (9216/16)
#define NW32  288       // u32 words per bit-packed row (576 u16)
#define THRESH 0.8f
#define MIN_GROUP 10

// ---------------------------------------------------------------------------
// Kernel 1: patch embed (4x4 stride-4 conv + bias) + row L2 normalize to f16.
// ---------------------------------------------------------------------------
__global__ void __launch_bounds__(128)
patch_embed_kernel(const float* __restrict__ x,
                   const float* __restrict__ W,
                   const float* __restrict__ b,
                   float* __restrict__ out,
                   _Float16* __restrict__ nrm) {
  const int n = blockIdx.x;          // token id
  const int c = threadIdx.x;         // output channel 0..127
  const int py = n / 96, px = n % 96;

  __shared__ float xs[48];
  __shared__ float wsum[4];

  if (c < 48) {
    const int ci = c >> 4, r = (c >> 2) & 3, cc = c & 3;
    xs[c] = x[ci * (384 * 384) + (py * 4 + r) * 384 + (px * 4 + cc)];
  }
  __syncthreads();

  float acc = b[c];
  const float* wc = W + c * 48;      // OIHW: [128][3][4][4]
#pragma unroll
  for (int i = 0; i < 48; ++i) acc += xs[i] * wc[i];

  float v = acc * acc;
#pragma unroll
  for (int off = 16; off > 0; off >>= 1) v += __shfl_xor(v, off, 32);
  const int lane = threadIdx.x & 31, wid = threadIdx.x >> 5;
  if (lane == 0) wsum[wid] = v;
  __syncthreads();
  const float total = wsum[0] + wsum[1] + wsum[2] + wsum[3];
  const float rn = rsqrtf(total);

  out[(size_t)n * EMB + c] = acc;
  nrm[(size_t)n * EMB + c] = (_Float16)(acc * rn);
}

// ---------------------------------------------------------------------------
// TDM: DMA one contiguous 8 KB panel (32 rows x 128 f16) from global to LDS.
// D# per CDNA5 ISA: 2D, data_size=8B, tile_dim0=1024.
// ---------------------------------------------------------------------------
__device__ __forceinline__ void tdm_load_panel_8k(unsigned long long ga,
                                                  unsigned lds_off) {
  u32x4 g0;
  g0[0] = 1u;                                    // count=1, user mode
  g0[1] = lds_off;                               // lds_addr[31:0]
  g0[2] = (unsigned)(ga & 0xFFFFFFFFull);        // global_addr[31:0]
  g0[3] = (unsigned)((ga >> 32) & 0x1FFFFFFull)  // global_addr[56:32]
          | 0x80000000u;                         // type=2 ("image")
  i32x8 g1;
  g1[0] = 0x30000;              // wg_mask=0 (not in cluster), data_size=3 (8B)
  g1[1] = (int)(1024u << 16);   // tensor_dim0[15:0]=1024 (barrier_addr=0)
  g1[2] = (int)(1u << 16);      // tensor_dim0[31:16]=0 | tensor_dim1[15:0]=1
  g1[3] = (int)(1024u << 16);   // tensor_dim1[31:16]=0 | tile_dim0=1024
  g1[4] = 1;                    // tile_dim1=1, tile_dim2=0
  g1[5] = 1024;                 // tensor_dim0_stride[31:0]=1024
  g1[6] = 0;                    // stride hi / dim1_stride lo
  g1[7] = 0;
  i32x4 g2 = {0, 0, 0, 0};      // dims 2..3 unused
  i32x4 g3 = {0, 0, 0, 0};      // dim 4 unused
  i32x8 g4 = {0, 0, 0, 0, 0, 0, 0, 0};  // trailing group (6-arg toolchain form)
  __builtin_amdgcn_tensor_load_to_lds(g0, g1, g2, g3, g4, 0);
}

// A-fragment (16-bit A 16x32, contiguous k = 16*hi + s, same order as B) from an LDS row pointer.
__device__ __forceinline__ v16h load_afrag(const _Float16* row, int kb, int hi) {
  v8h lo  = *(const v8h*)(row + kb + hi * 16);
  v8h hi8 = *(const v8h*)(row + kb + hi * 16 + 8);
  v16h a;
#pragma unroll
  for (int e = 0; e < 8; ++e) { a[e] = lo[e]; a[e + 8] = hi8[e]; }
  return a;
}

// Threshold one 16x16 f32 D tile and bit-pack 16 columns into u16 per row.
__device__ __forceinline__ void store_tile(const v8f& d, int rowbase, int jt,
                                           unsigned short* __restrict__ cb,
                                           int lane) {
  unsigned short bits = 0;
#pragma unroll
  for (int r = 0; r < 8; ++r) {
    unsigned long long m = __ballot(d[r] > THRESH);
    if (lane == r)     bits = (unsigned short)(m & 0xFFFFull);
    if (lane == r + 8) bits = (unsigned short)((m >> 16) & 0xFFFFull);
  }
  if (lane < 16)
    cb[(size_t)(rowbase + lane) * NT + jt] = bits;
}

// ---------------------------------------------------------------------------
// Kernel 2: cos_sim = nrm @ nrm^T, thresholded + bit-packed.
// 8 waves/block; block computes a 32x256 region. A panel (32x128) TDM-loaded
// once by wave 0; each wave TDM-loads its own 32-row B panel. 2x2 register
// blocking -> 16 v_wmma_f32_16x16x32_f16 per wave, all operands from LDS.
// ---------------------------------------------------------------------------
__global__ void __launch_bounds__(256)
cos_wmma_kernel(const _Float16* __restrict__ nrm,
                unsigned short* __restrict__ cb) {
  __shared__ _Float16 shA[32 * 128];        // 8 KB
  __shared__ _Float16 shB[8][32 * 128];     // 64 KB

  const int tid  = threadIdx.x;
  const int lane = tid & 31;
  const int wv   = __builtin_amdgcn_readfirstlane(tid >> 5);  // wave id 0..7
  const int ibase = blockIdx.y * 32;
  const int jpair = blockIdx.x * 8 + wv;    // 0..287
  const int jbase = jpair * 32;

  // --- stage panels via Tensor Data Mover ---
  const unsigned ldsA = (unsigned)(size_t)(void*)shA;
  const unsigned ldsB = (unsigned)(size_t)(void*)&shB[wv][0];
  const unsigned long long gaA =
      (unsigned long long)(size_t)(const void*)(nrm + (size_t)ibase * EMB);
  const unsigned long long gaB =
      (unsigned long long)(size_t)(const void*)(nrm + (size_t)jbase * EMB);

  if (wv == 0) tdm_load_panel_8k(gaA, ldsA);  // wave-uniform branch
  tdm_load_panel_8k(gaB, ldsB);
  __builtin_amdgcn_s_wait_tensorcnt(0);       // own DMAs done
  __syncthreads();                            // publish A panel to all waves

  const int l  = lane & 15;
  const int hi = lane >> 4;
  const _Float16* Ab = shA;
  const _Float16* Bb = &shB[wv][0];

  v8f d00 = {}, d01 = {}, d10 = {}, d11 = {};
#pragma unroll
  for (int kk = 0; kk < 4; ++kk) {
    const int kb = kk * 32;
    // A 16x32 frags for the two i-tiles (ds_load_b128 x4)
    v16h a0 = load_afrag(Ab + (size_t)(0 + l) * EMB, kb, hi);
    v16h a1 = load_afrag(Ab + (size_t)(16 + l) * EMB, kb, hi);
    // B 32x16 frags: B column == nrm row -> 16 contiguous halves per lane
    v16h b0 = *(const v16h*)(Bb + (size_t)(0 + l) * EMB + kb + hi * 16);
    v16h b1 = *(const v16h*)(Bb + (size_t)(16 + l) * EMB + kb + hi * 16);
    d00 = __builtin_amdgcn_wmma_f32_16x16x32_f16(false, a0, false, b0,
                                                 (short)0, d00, false, false);
    d01 = __builtin_amdgcn_wmma_f32_16x16x32_f16(false, a0, false, b1,
                                                 (short)0, d01, false, false);
    d10 = __builtin_amdgcn_wmma_f32_16x16x32_f16(false, a1, false, b0,
                                                 (short)0, d10, false, false);
    d11 = __builtin_amdgcn_wmma_f32_16x16x32_f16(false, a1, false, b1,
                                                 (short)0, d11, false, false);
  }

  const int jt = jpair * 2;
  store_tile(d00, ibase,      jt,     cb, lane);
  store_tile(d01, ibase,      jt + 1, cb, lane);
  store_tile(d10, ibase + 16, jt,     cb, lane);
  store_tile(d11, ibase + 16, jt + 1, cb, lane);
}

// ---------------------------------------------------------------------------
// Kernel 3: degrees s[i] = popcount(row i), rebuild_idx init = arange.
// ---------------------------------------------------------------------------
__global__ void __launch_bounds__(256)
degree_init_kernel(const unsigned short* __restrict__ cb,
                   int* __restrict__ s, int* __restrict__ ridx) {
  const int lane = threadIdx.x & 31;
  const int row  = blockIdx.x * 8 + (threadIdx.x >> 5);
  if (row >= N_TOK) return;
  const unsigned* crow = (const unsigned*)(cb + (size_t)row * NT);
  int cnt = 0;
  for (int w = lane; w < NW32; w += 32) cnt += __popc(crow[w]);
#pragma unroll
  for (int off = 16; off > 0; off >>= 1) cnt += __shfl_xor(cnt, off, 32);
  if (lane == 0) { s[row] = cnt; ridx[row] = row; }
}

// ---------------------------------------------------------------------------
// Kernel 4: greedy clustering while-loop (single persistent workgroup:
// device-side global sync between data-dependent iterations).
// ---------------------------------------------------------------------------
__global__ void __launch_bounds__(1024)
cluster_loop_kernel(unsigned short* __restrict__ cb,
                    int* __restrict__ s, int* __restrict__ ridx) {
  const int tid  = threadIdx.x;
  const int lane = tid & 31;
  const int wid  = tid >> 5;

  __shared__ unsigned rowmask[NW32];
  __shared__ int redV[32];
  __shared__ int redI[32];
  __shared__ int bcastV, bcastI;

  for (;;) {
    // ---- argmax(s), first index on ties (matches jnp.argmax) ----
    int bv = -1, bi = 0;
    for (int i = tid; i < N_TOK; i += 1024) {
      const int v = s[i];
      if (v > bv) { bv = v; bi = i; }
    }
#pragma unroll
    for (int off = 16; off > 0; off >>= 1) {
      const int ov = __shfl_xor(bv, off, 32);
      const int oi = __shfl_xor(bi, off, 32);
      if (ov > bv || (ov == bv && oi < bi)) { bv = ov; bi = oi; }
    }
    if (lane == 0) { redV[wid] = bv; redI[wid] = bi; }
    __syncthreads();
    if (tid == 0) {
      int fv = redV[0], fi = redI[0];
      for (int w = 1; w < 32; ++w)
        if (redV[w] > fv || (redV[w] == fv && redI[w] < fi)) {
          fv = redV[w]; fi = redI[w];
        }
      bcastV = fv; bcastI = fi;
    }
    __syncthreads();
    const int maxv = bcastV;
    const int t    = bcastI;
    if (maxv <= MIN_GROUP) break;

    const unsigned* trow = (const unsigned*)(cb + (size_t)t * NT);
    for (int w = tid; w < NW32; w += 1024) rowmask[w] = trow[w];
    __syncthreads();

    for (int j = tid; j < N_TOK; j += 1024)
      if ((rowmask[j >> 5] >> (j & 31)) & 1u) ridx[j] = t;

    for (int row = wid; row < N_TOK; row += 32) {
      unsigned* crow = (unsigned*)(cb + (size_t)row * NT);
      int rem = 0;
      for (int w = lane; w < NW32; w += 32) {
        const unsigned c    = crow[w];
        const unsigned kill = c & rowmask[w];
        if (kill) { rem += __popc(kill); crow[w] = c & ~rowmask[w]; }
      }
#pragma unroll
      for (int off = 16; off > 0; off >>= 1) rem += __shfl_xor(rem, off, 32);
      if (lane == 0 && rem) s[row] -= rem;
    }
    __syncthreads();
  }
}

// ---------------------------------------------------------------------------
// Kernel 5: emit rebuild_idx and centroid_mask as floats (tuple tail).
// ---------------------------------------------------------------------------
__global__ void __launch_bounds__(256)
finalize_kernel(const int* __restrict__ ridx,
                float* __restrict__ out_ridx,
                float* __restrict__ out_mask) {
  const int i = blockIdx.x * 256 + threadIdx.x;
  if (i >= N_TOK) return;
  const int r = ridx[i];
  out_ridx[i] = (float)r;
  out_mask[i] = (r == i) ? 1.0f : 0.0f;
}

// ---------------------------------------------------------------------------
extern "C" void kernel_launch(void* const* d_in, const int* in_sizes, int n_in,
                              void* d_out, int out_size, void* d_ws, size_t ws_size,
                              hipStream_t stream) {
  const float* x = (const float*)d_in[0];   // [1,3,384,384]
  const float* W = (const float*)d_in[1];   // [128,3,4,4]
  const float* b = (const float*)d_in[2];   // [128]
  float* out = (float*)d_out;               // out | rebuild_idx | centroid_mask

  // workspace layout (bytes):
  //   nrm  f16 : 2,359,296 | cb u16 : 10,616,832 | s i32 : 36,864 | ridx i32
  char* ws = (char*)d_ws;
  _Float16*       nrm  = (_Float16*)ws;
  unsigned short* cb   = (unsigned short*)(ws + 2359296);
  int*            s    = (int*)(ws + 2359296 + 10616832);
  int*            ridx = (int*)(ws + 2359296 + 10616832 + 36864);

  patch_embed_kernel<<<N_TOK, 128, 0, stream>>>(x, W, b, out, nrm);

  dim3 g2(NT / 16, NT / 2);   // 36 x 288 blocks, 8 waves each, 32x256 region
  cos_wmma_kernel<<<g2, 256, 0, stream>>>(nrm, cb);

  degree_init_kernel<<<N_TOK / 8, 256, 0, stream>>>(cb, s, ridx);

  cluster_loop_kernel<<<1, 1024, 0, stream>>>(cb, s, ridx);

  finalize_kernel<<<(N_TOK + 255) / 256, 256, 0, stream>>>(
      ridx, out + (size_t)N_TOK * EMB, out + (size_t)N_TOK * EMB + N_TOK);
}
